// GPT_86277303042392
// MI455X (gfx1250) — matrix operand order repaired and checked by reading, not compiled, not run on hardware
//
#include <hip/hip_runtime.h>
#include <hip/hip_bf16.h>
#include <math.h>

// ---------------------------------------------------------------------------
// GPT-2 small forward (B=2,T=1024,E=768,H=12,HD=64,L=12,V=50257) for gfx1250.
// All GEMMs via v_wmma_f32_16x16x32_bf16; bf16 operand chain end-to-end
// (fp32 residual stream). GEMM B-panels staged into LDS with the Tensor Data
// Mover (tensor_load_to_lds + s_wait_tensorcnt), wave tile 32x32 (4 accs).
// ---------------------------------------------------------------------------

#define GPT_B   2
#define GPT_T   1024
#define GPT_E   768
#define GPT_H   12
#define GPT_HD  64
#define GPT_L   12
#define GPT_V   50257
#define GPT_M   (GPT_B * GPT_T)      // 2048 activation rows
#define GPT_3E  (3 * GPT_E)          // 2304

typedef __bf16 bf16_t;
typedef bf16_t v16bf __attribute__((ext_vector_type(16)));
typedef float  v8f   __attribute__((ext_vector_type(8)));
typedef unsigned int u32x4 __attribute__((ext_vector_type(4)));
typedef int          i32x8 __attribute__((ext_vector_type(8)));
typedef int          i32x4 __attribute__((ext_vector_type(4)));

__device__ __forceinline__ v8f wmma_bf16(v16bf a, v16bf b, v8f c) {
  // (neg_a, A, neg_b, B, c_mod, C, reuse_a, reuse_b)
  return __builtin_amdgcn_wmma_f32_16x16x32_bf16(false, a, false, b,
                                                 (short)0, c, false, false);
}

// k index inside a 16x32 bf16 A-fragment for element e of lane-half `half`
// (contiguous order matching B: lanes0-15 K0..15, lanes16-31 K16..31)
__device__ __forceinline__ int a_kidx(int e, int half) {
  return e + 16 * half;
}

// ---------------------------------------------------------------------------
// Tensor Data Mover: DMA a 2D bf16 tile (rows x rowlen, row stride = rowlen)
// from global memory into LDS. D# per CDNA5 ISA 8.3-8.6:
//   group0: count=1 | lds_addr | global_addr[56:0] | type=2
//   group1: data_size=1(2B) | tensor_dim0/1 | tile_dim0/1 | dim0_stride
// Issued by one wave (TDM ignores EXEC); completion via TENSORcnt.
// This toolchain exposes the 6-arg builtin:
//   (u32x4 g0, i32x8 g1, i32x4 g2, i32x4 g3, i32x8 pad, i32 cpol)
// ---------------------------------------------------------------------------
#if __has_builtin(__builtin_amdgcn_tensor_load_to_lds)
#define HAVE_TDM 1
__device__ __forceinline__ void tdm_load_2d_bf16(
    unsigned lds_addr, const void* gaddr, int rows, int rowlen, int tensor_rows) {
  unsigned long long ga = (unsigned long long)gaddr;
  u32x4 g0;
  g0[0] = 1u;                                        // count=1, user mode
  g0[1] = lds_addr;                                  // LDS byte address
  g0[2] = (unsigned)(ga & 0xffffffffu);              // global addr [31:0]
  g0[3] = (unsigned)((ga >> 32) & 0x01ffffffu)       // global addr [56:32]
          | (2u << 30);                              // type = 2 ("image")
  i32x8 g1;
  g1[0] = 1 << 16;                                   // data_size = 1 (2 bytes)
  g1[1] = (rowlen & 0xffff) << 16;                   // tensor_dim0 [15:0] @bit48
  g1[2] = ((rowlen >> 16) & 0xffff)                  // tensor_dim0 [31:16]
          | ((tensor_rows & 0xffff) << 16);          // tensor_dim1 [15:0]
  g1[3] = ((tensor_rows >> 16) & 0xffff)             // tensor_dim1 [31:16]
          | ((rowlen & 0xffff) << 16);               // tile_dim0
  g1[4] = rows & 0xffff;                             // tile_dim1 (tile_dim2=0)
  g1[5] = rowlen;                                    // tensor_dim0_stride [31:0]
  g1[6] = 0;                                         // stride msbs / dim1_stride
  g1[7] = 0;
  i32x4 z4 = {0, 0, 0, 0};
  i32x8 z8 = {0, 0, 0, 0, 0, 0, 0, 0};
  __builtin_amdgcn_tensor_load_to_lds(g0, g1, z4, z4, z8, 0);
}
#else
#define HAVE_TDM 0
#endif

// ---------------------------------------------------------------------------
// Embedding: x[b,t,:] = tok_emb[ids[b,t]] + pos_emb[t]   (fp32 residual)
// ---------------------------------------------------------------------------
__global__ __launch_bounds__(256) void embed_kernel(
    const int* __restrict__ ids, const float* __restrict__ tok,
    const float* __restrict__ pos, float* __restrict__ x) {
  int i = blockIdx.x * 256 + threadIdx.x;
  if (i >= GPT_M * GPT_E) return;
  int e  = i % GPT_E;
  int bt = i / GPT_E;
  int t  = bt % GPT_T;
  x[i] = tok[(size_t)ids[bt] * GPT_E + e] + pos[(size_t)t * GPT_E + e];
}

// ---------------------------------------------------------------------------
// LayerNorm over rows of 768, fp32 in -> bf16 out (GEMM-ready A operand)
// ---------------------------------------------------------------------------
__global__ __launch_bounds__(256) void layernorm_bf16_kernel(
    const float* __restrict__ X, const float* __restrict__ sc,
    const float* __restrict__ bi, bf16_t* __restrict__ Y) {
  __shared__ float red[256];
  __shared__ float stats[2];
  int row = blockIdx.x, tid = threadIdx.x;
  const float* x = X + (size_t)row * GPT_E;
  float v0 = x[tid], v1 = x[tid + 256], v2 = x[tid + 512];
  red[tid] = v0 + v1 + v2;
  __syncthreads();
  for (int o = 128; o > 0; o >>= 1) {
    if (tid < o) red[tid] += red[tid + o];
    __syncthreads();
  }
  if (tid == 0) stats[0] = red[0] * (1.0f / GPT_E);
  __syncthreads();
  float m = stats[0];
  float d0 = v0 - m, d1 = v1 - m, d2 = v2 - m;
  red[tid] = d0 * d0 + d1 * d1 + d2 * d2;
  __syncthreads();
  for (int o = 128; o > 0; o >>= 1) {
    if (tid < o) red[tid] += red[tid + o];
    __syncthreads();
  }
  if (tid == 0) stats[1] = rsqrtf(red[0] * (1.0f / GPT_E) + 1e-5f);
  __syncthreads();
  float rs = stats[1];
  bf16_t* y = Y + (size_t)row * GPT_E;
  y[tid]       = (bf16_t)(d0 * rs * sc[tid]       + bi[tid]);
  y[tid + 256] = (bf16_t)(d1 * rs * sc[tid + 256] + bi[tid + 256]);
  y[tid + 512] = (bf16_t)(d2 * rs * sc[tid + 512] + bi[tid + 512]);
}

// ---------------------------------------------------------------------------
// Transpose-convert one 768x768 fp32 weight W[k][n] -> bf16 Wt[n][k]
// ---------------------------------------------------------------------------
__global__ __launch_bounds__(256) void transpose_bf16_kernel(
    const float* __restrict__ src, bf16_t* __restrict__ dst) {
  __shared__ float tile[32][33];
  int bx = blockIdx.x * 32;  // n base
  int by = blockIdx.y * 32;  // k base
  int tx = threadIdx.x;      // 0..31
  int ty = threadIdx.y;      // 0..7
  for (int i = ty; i < 32; i += 8)
    tile[i][tx] = src[(size_t)(by + i) * GPT_E + (bx + tx)];
  __syncthreads();
  for (int i = ty; i < 32; i += 8)
    dst[(size_t)(bx + i) * GPT_E + (by + tx)] = (bf16_t)tile[tx][i];
}

// ---------------------------------------------------------------------------
// v-transpose: qkv bf16 [B*T, 3E] (v block) -> vT bf16 [B,H,HD,T]
// ---------------------------------------------------------------------------
__global__ __launch_bounds__(256) void vtranspose_kernel(
    const bf16_t* __restrict__ qkv, bf16_t* __restrict__ vT) {
  __shared__ bf16_t tile[32][33];
  int t0 = blockIdx.x * 32;
  int d0 = blockIdx.y * 32;
  int bh = blockIdx.z;                       // b*H + h
  int b  = bh / GPT_H, h = bh % GPT_H;
  int tx = threadIdx.x, ty = threadIdx.y;
  const bf16_t* src = qkv + (size_t)b * GPT_T * GPT_3E + 2 * GPT_E + h * GPT_HD;
  for (int i = ty; i < 32; i += 8)
    tile[i][tx] = src[(size_t)(t0 + i) * GPT_3E + (d0 + tx)];
  __syncthreads();
  bf16_t* dst = vT + (size_t)bh * GPT_HD * GPT_T;
  for (int i = ty; i < 32; i += 8)
    dst[(size_t)(d0 + i) * GPT_T + (t0 + tx)] = tile[tx][i];
}

// ---------------------------------------------------------------------------
// WMMA GEMM: C[M,N] = A[M,K] @ B[K,N] (+ epilogue)
//   A: bf16 row-major. Bt: B transposed [N,K]:
//     bf16 path (weights, K=768): 64-row B panel DMA'd into LDS by the TDM,
//       fragments read via ds_load.
//     fp32 path (tied logits head, tok_emb [V,E]): direct global + convert.
//   epi: 0=none  2=+bias+fp32 residual  3=gelu(+bias)
//   out_bf16: store bf16 (activations) or fp32 (residual stream / logits)
// Block 128 thr = 4 waves in 2x2; wave tile 32x32; block tile 64x64.
// ---------------------------------------------------------------------------
__global__ __launch_bounds__(128) void gemm_wmma_kernel(
    const bf16_t* __restrict__ A, const void* __restrict__ Bt,
    const float* __restrict__ bias, const float* __restrict__ Res,
    void* __restrict__ Cout, int M, int N, int K,
    int epi, int b_is_f32, int out_bf16) {
  __shared__ bf16_t Bs[64 * GPT_E];               // 96 KB B panel (bf16 path)
  int lane = threadIdx.x & 31;
  int wave = threadIdx.x >> 5;                    // 0..3
  int half = lane >> 4;
  int mb   = blockIdx.y * 64 + (wave >> 1) * 32;  // wave m-base
  int nb   = blockIdx.x * 64 + (wave & 1) * 32;   // wave n-base
  int am0  = mb + (lane & 15), am1 = am0 + 16;
  int bn0  = nb + (lane & 15), bn1 = bn0 + 16;

  const bf16_t* arow0 = A + (size_t)am0 * K;
  const bf16_t* arow1 = A + (size_t)am1 * K;
  v8f acc00 = {}, acc01 = {}, acc10 = {}, acc11 = {};

  if (b_is_f32) {
    int bn0c = (bn0 < N) ? bn0 : (N - 1);
    int bn1c = (bn1 < N) ? bn1 : (N - 1);
    const float* br0 = (const float*)Bt + (size_t)bn0c * K + 16 * half;
    const float* br1 = (const float*)Bt + (size_t)bn1c * K + 16 * half;
    for (int kb = 0; kb < K; kb += 32) {
      __builtin_prefetch(br0 + kb + 32, 0, 0);
      __builtin_prefetch(br1 + kb + 32, 0, 0);
      v16bf a0, a1, b0, b1;
#pragma unroll
      for (int e = 0; e < 16; ++e) {
        int ki = kb + a_kidx(e, half);
        a0[e] = arow0[ki];
        a1[e] = arow1[ki];
        b0[e] = (bf16_t)br0[kb + e];
        b1[e] = (bf16_t)br1[kb + e];
      }
      acc00 = wmma_bf16(a0, b0, acc00);
      acc01 = wmma_bf16(a0, b1, acc01);
      acc10 = wmma_bf16(a1, b0, acc10);
      acc11 = wmma_bf16(a1, b1, acc11);
    }
  } else {
    // ---- stage the 64 x K bf16 B panel into LDS ----
    const bf16_t* bpanel = (const bf16_t*)Bt + (size_t)(blockIdx.x * 64) * K;
#if HAVE_TDM
    if (wave == 0) {
      tdm_load_2d_bf16((unsigned)(size_t)(void*)Bs, bpanel, 64, K, N);
      __builtin_amdgcn_s_wait_tensorcnt(0);
    }
#else
    {
      const uint4* gsrc = (const uint4*)bpanel;          // 8 bf16 per uint4
      uint4* ldst = (uint4*)Bs;
      int nvec = 64 * K / 8;
      for (int i = threadIdx.x; i < nvec; i += 128) ldst[i] = gsrc[i];
    }
#endif
    __syncthreads();

    const bf16_t* br0 =
        Bs + (size_t)((wave & 1) * 32 + (lane & 15)) * K + 16 * half;
    const bf16_t* br1 = br0 + (size_t)16 * K;
    for (int kb = 0; kb < K; kb += 32) {
      __builtin_prefetch(arow0 + kb + 32, 0, 0);
      __builtin_prefetch(arow1 + kb + 32, 0, 0);
      v16bf a0, a1, b0, b1;
#pragma unroll
      for (int e = 0; e < 16; ++e) {
        int ki = kb + a_kidx(e, half);
        a0[e] = arow0[ki];
        a1[e] = arow1[ki];
        b0[e] = br0[kb + e];
        b1[e] = br1[kb + e];
      }
      acc00 = wmma_bf16(a0, b0, acc00);
      acc01 = wmma_bf16(a0, b1, acc01);
      acc10 = wmma_bf16(a1, b0, acc10);
      acc11 = wmma_bf16(a1, b1, acc11);
    }
  }

  // ---- epilogue: 4 16x16 tiles ----
  int mrow = mb + (half << 3);
#pragma unroll
  for (int tile = 0; tile < 4; ++tile) {
    v8f acc = (tile == 0) ? acc00 : (tile == 1) ? acc01
            : (tile == 2) ? acc10 : acc11;
    int n   = (tile & 1) ? bn1 : bn0;
    int m0  = mrow + ((tile >> 1) << 4);
    if (n < N) {
      float bv = (bias != nullptr) ? bias[n] : 0.0f;
#pragma unroll
      for (int r = 0; r < 8; ++r) {
        float v = acc[r] + bv;
        if (epi == 3) v = 0.5f * v * (1.0f + erff(v * 0.70710678118654752f));
        if (epi == 2) v += Res[(size_t)(m0 + r) * N + n];
        if (out_bf16) ((bf16_t*)Cout)[(size_t)(m0 + r) * N + n] = (bf16_t)v;
        else          ((float*)Cout)[(size_t)(m0 + r) * N + n] = v;
      }
    }
  }
}

// ---------------------------------------------------------------------------
// Attention for one (b, h, 16-row t-tile), all-bf16 operands.
//   scores[t,s] = scale*dot(k[t],q[s]); causal keep s<=t; softmax over s;
//   attn[t,d] = sum_s w[t,s] v[s,d]  (v read via vT for contiguous loads).
// Block 128 thr = 4 waves. LDS: 64KB fp32 scores + 32KB bf16 weights.
// ---------------------------------------------------------------------------
__global__ __launch_bounds__(128) void attn_kernel(
    const bf16_t* __restrict__ qkv, const bf16_t* __restrict__ vT,
    bf16_t* __restrict__ attn) {
  __shared__ float  S[16 * GPT_T];    // 64 KB scores
  __shared__ bf16_t Wb[16 * GPT_T];   // 32 KB softmax weights

  int tb   = blockIdx.x;              // t-tile (0..63)
  int h    = blockIdx.y;
  int bB   = blockIdx.z;
  int lane = threadIdx.x & 31;
  int wave = threadIdx.x >> 5;
  int half = lane >> 4;

  const bf16_t* qb  = qkv + (size_t)bB * GPT_T * GPT_3E + h * GPT_HD;
  const bf16_t* kbp = qb + GPT_E;
  const float scale = 0.125f;         // 1/sqrt(64)

  // cache the two k-tile A-fragments (invariant over all 64 n-tiles)
  v16bf ak0, ak1;
  {
    int m = tb * 16 + (lane & 15);
    const bf16_t* r = kbp + (size_t)m * GPT_3E;
#pragma unroll
    for (int e = 0; e < 16; ++e) {
      ak0[e] = r[a_kidx(e, half)];
      ak1[e] = r[32 + a_kidx(e, half)];
    }
  }

  // scores: 16 x 1024, each wave 16 n-tiles
  for (int nt = wave; nt < GPT_T / 16; nt += 4) {
    int s = nt * 16 + (lane & 15);
    const bf16_t* qr = qb + (size_t)s * GPT_3E + 16 * half;
    v16bf b0, b1;
#pragma unroll
    for (int e = 0; e < 16; ++e) { b0[e] = qr[e]; b1[e] = qr[32 + e]; }
    v8f acc = {};
    acc = wmma_bf16(ak0, b0, acc);
    acc = wmma_bf16(ak1, b1, acc);
    int n  = nt * 16 + (lane & 15);
    int m0 = half << 3;
#pragma unroll
    for (int r = 0; r < 8; ++r) S[(m0 + r) * GPT_T + n] = acc[r] * scale;
  }
  __syncthreads();

  // causal softmax: wave owns 4 rows, lane-strided + wave32 shuffles
  for (int rr = 0; rr < 4; ++rr) {
    int row = wave * 4 + rr;
    int tg  = tb * 16 + row;          // keep s <= tg
    float mx = -1e30f;
    for (int s = lane; s <= tg; s += 32) mx = fmaxf(mx, S[row * GPT_T + s]);
    for (int o = 16; o > 0; o >>= 1) mx = fmaxf(mx, __shfl_xor(mx, o, 32));
    float sum = 0.0f;
    for (int s = lane; s <= tg; s += 32) {
      float ev = __expf(S[row * GPT_T + s] - mx);
      S[row * GPT_T + s] = ev;
      sum += ev;
    }
    for (int o = 16; o > 0; o >>= 1) sum += __shfl_xor(sum, o, 32);
    float inv = 1.0f / sum;
    for (int s = lane; s < GPT_T; s += 32) {
      float wv = (s <= tg) ? S[row * GPT_T + s] * inv : 0.0f;
      Wb[row * GPT_T + s] = (bf16_t)wv;
    }
  }
  __syncthreads();

  // attn = w @ v : 16 x 64, wave owns one d-tile; vT rows contiguous
  v8f acc = {};
  {
    int d = wave * 16 + (lane & 15);
    const bf16_t* vrow =
        vT + ((size_t)(bB * GPT_H + h) * GPT_HD + d) * GPT_T + 16 * half;
    const bf16_t* wrow = Wb + (lane & 15) * GPT_T;
    for (int kk = 0; kk < GPT_T; kk += 32) {
      v16bf a, b;
#pragma unroll
      for (int e = 0; e < 16; ++e) {
        a[e] = wrow[kk + a_kidx(e, half)];
        b[e] = vrow[kk + e];
      }
      acc = wmma_bf16(a, b, acc);
    }
  }
  {
    int d  = wave * 16 + (lane & 15);
    int m0 = half << 3;
    bf16_t* out = attn + (size_t)(bB * GPT_T + tb * 16) * GPT_E + h * GPT_HD;
#pragma unroll
    for (int r = 0; r < 8; ++r)
      out[(size_t)(m0 + r) * GPT_E + d] = (bf16_t)acc[r];
  }
}

// ---------------------------------------------------------------------------
// Host-side orchestration
// ---------------------------------------------------------------------------
extern "C" void kernel_launch(void* const* d_in, const int* in_sizes, int n_in,
                              void* d_out, int out_size, void* d_ws, size_t ws_size,
                              hipStream_t stream) {
  const int*   ids   = (const int*)d_in[0];
  const float* tok   = (const float*)d_in[1];
  const float* pos   = (const float*)d_in[2];
  const float* ln1s  = (const float*)d_in[3];
  const float* ln1b  = (const float*)d_in[4];
  const float* Wq    = (const float*)d_in[5];
  const float* Wk    = (const float*)d_in[6];
  const float* Wv    = (const float*)d_in[7];
  const float* Wp    = (const float*)d_in[8];
  const float* bp    = (const float*)d_in[9];
  const float* ln2s  = (const float*)d_in[10];
  const float* ln2b  = (const float*)d_in[11];
  const float* W1    = (const float*)d_in[12];
  const float* b1    = (const float*)d_in[13];
  const float* W2    = (const float*)d_in[14];
  const float* b2    = (const float*)d_in[15];
  const float* lnfs  = (const float*)d_in[16];
  const float* lnfb  = (const float*)d_in[17];
  float* logits = (float*)d_out;

  // workspace carve-up (~35 MB): fp32 residual + bf16 everything else
  float*  x      = (float*)d_ws;
  bf16_t* h      = (bf16_t*)(x + (size_t)GPT_M * GPT_E);
  bf16_t* qkv    = h      + (size_t)GPT_M * GPT_E;
  bf16_t* vT     = qkv    + (size_t)GPT_M * GPT_3E;
  bf16_t* attn   = vT     + (size_t)GPT_B * GPT_H * GPT_HD * GPT_T;
  bf16_t* y1     = attn   + (size_t)GPT_M * GPT_E;
  bf16_t* wt_qkv = y1     + (size_t)GPT_M * GPT_E;
  bf16_t* wt_p   = wt_qkv + (size_t)GPT_3E * GPT_E;
  bf16_t* wt_1   = wt_p   + (size_t)GPT_E * GPT_E;
  bf16_t* wt_2   = wt_1   + (size_t)GPT_E * GPT_E;

  dim3 tpbT(32, 8), grdT(GPT_E / 32, GPT_E / 32);
  dim3 grdV(GPT_T / 32, GPT_HD / 32, GPT_B * GPT_H);
  const size_t EE = (size_t)GPT_E * GPT_E;

  embed_kernel<<<(GPT_M * GPT_E + 255) / 256, 256, 0, stream>>>(ids, tok, pos, x);

  for (int l = 0; l < GPT_L; ++l) {
    // repack weights: [K,N] fp32 -> [N,K] bf16 (QKV packed into one Bt)
    transpose_bf16_kernel<<<grdT, tpbT, 0, stream>>>(Wq + l * EE, wt_qkv);
    transpose_bf16_kernel<<<grdT, tpbT, 0, stream>>>(Wk + l * EE, wt_qkv + EE);
    transpose_bf16_kernel<<<grdT, tpbT, 0, stream>>>(Wv + l * EE, wt_qkv + 2 * EE);
    transpose_bf16_kernel<<<grdT, tpbT, 0, stream>>>(Wp + l * EE, wt_p);
    transpose_bf16_kernel<<<grdT, tpbT, 0, stream>>>(W1 + l * EE, wt_1);
    transpose_bf16_kernel<<<grdT, tpbT, 0, stream>>>(W2 + l * EE, wt_2);

    // h = ln1(x)  [bf16]
    layernorm_bf16_kernel<<<GPT_M, 256, 0, stream>>>(
        x, ln1s + l * GPT_E, ln1b + l * GPT_E, h);
    // qkv = h @ [Wq|Wk|Wv]  [bf16 out, no bias]
    gemm_wmma_kernel<<<dim3(GPT_3E / 64, GPT_M / 64), 128, 0, stream>>>(
        h, wt_qkv, nullptr, nullptr, qkv, GPT_M, GPT_3E, GPT_E, 0, 0, 1);
    // vT for contiguous w@v loads
    vtranspose_kernel<<<grdV, tpbT, 0, stream>>>(qkv, vT);
    // attention -> attn [bf16]
    attn_kernel<<<dim3(GPT_T / 16, GPT_H, GPT_B), 128, 0, stream>>>(qkv, vT, attn);
    // x = x + attn @ Wp + bp  [fp32 out]
    gemm_wmma_kernel<<<dim3(GPT_E / 64, GPT_M / 64), 128, 0, stream>>>(
        attn, wt_p, bp + l * GPT_E, x, x, GPT_M, GPT_E, GPT_E, 2, 0, 0);
    // h = ln2(x)  [bf16]
    layernorm_bf16_kernel<<<GPT_M, 256, 0, stream>>>(
        x, ln2s + l * GPT_E, ln2b + l * GPT_E, h);
    // y1 = gelu(h @ W1 + b1)  [bf16 out]
    gemm_wmma_kernel<<<dim3(GPT_E / 64, GPT_M / 64), 128, 0, stream>>>(
        h, wt_1, b1 + l * GPT_E, nullptr, y1, GPT_M, GPT_E, GPT_E, 3, 0, 1);
    // x = x + y1 @ W2 + b2  [fp32 out]
    gemm_wmma_kernel<<<dim3(GPT_E / 64, GPT_M / 64), 128, 0, stream>>>(
        y1, wt_2, b2 + l * GPT_E, x, x, GPT_M, GPT_E, GPT_E, 2, 0, 0);
  }

  // final LN + tied logits head (tok_emb is [V,E] = Bt layout, fp32 source)
  layernorm_bf16_kernel<<<GPT_M, 256, 0, stream>>>(x, lnfs, lnfb, h);
  gemm_wmma_kernel<<<dim3((GPT_V + 63) / 64, GPT_M / 64), 128, 0, stream>>>(
      h, tok, nullptr, nullptr, logits, GPT_M, GPT_V, GPT_E, 0, 1, 0);
}
